// EntAttr_Model_5403068859161
// MI455X (gfx1250) — compile-verified, matrix operand order checked by reading, not run on hardware
//
#include <hip/hip_runtime.h>

#define HIDDEN  256
#define KDIM    768
#define MDIM    1024
#define NDIM    100000
#define BM      256
#define BN      128
#define BK      32
#define PITCH_A 40          // 32 data + 8 pad bf16 (20 dwords): conflict-free ds_load_b128
#define PITCH_B 40
#define NSTEP   (KDIM / BK) // 24

typedef __attribute__((ext_vector_type(16))) __bf16 v16bf;
typedef __attribute__((ext_vector_type(8)))  __bf16 bf16x8;
typedef __attribute__((ext_vector_type(4)))  __bf16 bf16x4;
typedef __attribute__((ext_vector_type(8)))  float  v8f;
typedef __attribute__((ext_vector_type(4)))  float  f32x4;

// ---------------------------------------------------------------------------
// Kernel 1: gather + pack A = bf16(concat(w1, w2, rel)) -> (1024, 768) in ws
// ---------------------------------------------------------------------------
__global__ __launch_bounds__(HIDDEN) void prep_kernel(
    const long long* __restrict__ batch_data,   // (1024, 2) i64
    const int*       __restrict__ ent2word,     // (100000, 2) i32
    const float*     __restrict__ words_embd,   // (50000, 256) f32
    const float*     __restrict__ rel_embed,    // (500, 256) f32
    __bf16*          __restrict__ A)            // (1024, 768) bf16
{
  const int row = blockIdx.x;
  const int t   = threadIdx.x;                  // 0..255
  const long long ent = batch_data[row * 2 + 0];
  const long long rl  = batch_data[row * 2 + 1];
  const int w1 = ent2word[ent * 2 + 0];
  const int w2 = ent2word[ent * 2 + 1];
  __bf16* dst = A + (size_t)row * KDIM;
  dst[t]              = (__bf16)words_embd[(size_t)w1 * HIDDEN + t];
  dst[HIDDEN + t]     = (__bf16)words_embd[(size_t)w2 * HIDDEN + t];
  dst[2 * HIDDEN + t] = (__bf16)rel_embed[(size_t)rl * HIDDEN + t];
}

// ---------------------------------------------------------------------------
// Kernel 2: out = A @ W^T + b via v_wmma_f32_16x16x32_bf16, LDS double-buffered
//   block: 256 threads = 8 wave32, tile 256(M) x 128(N), K step 32
//   A tile: async global->LDS (bf16, 16KB/step), ASYNCcnt-tracked
//   B tile: coalesced b128 f32 loads + cvt_pk_bf16 + ds_store (10KB/step)
//   inner loop: pure ds_load_b128 fragments + 16 WMMAs
// ---------------------------------------------------------------------------
__global__ __launch_bounds__(256) void gemm_kernel(
    const __bf16* __restrict__ A,      // (1024, 768) bf16
    const float*  __restrict__ W,      // (100000, 768) f32
    const float*  __restrict__ bias,   // (100000,) f32
    float*        __restrict__ out)    // (1024, 100000) f32
{
  __shared__ __bf16 As[2][BM * PITCH_A];   // 2 x 20 KB
  __shared__ __bf16 Bs[2][BN * PITCH_B];   // 2 x 10 KB

  const int tid      = threadIdx.x;
  const int wave     = tid >> 5;
  const int lane     = tid & 31;
  const int laneRow  = lane & 15;   // tile row (A: m, B: n)
  const int laneHalf = lane >> 4;   // K-half select
  const int waveM    = wave >> 1;   // 0..3
  const int waveN    = wave & 1;    // 0..1

  const int m0b = blockIdx.x * BM;
  const int n0b = blockIdx.y * BN;
  const int m0w = waveM * 64;       // wave offsets inside the block tile
  const int n0w = waveN * 64;

  v8f acc[4][4] = {};

  // ---- stage one K-step (BK=32) of A (async) and B (sync+cvt) into buf ----
  auto stage = [&](int buf, int k0) {
    // A: 256x32 bf16 = 1024 16B chunks, 4 per thread, async to LDS
#pragma unroll
    for (int q = 0; q < 4; ++q) {
      const int c  = q * 256 + tid;
      const int r  = c >> 2;        // row 0..255 (4 chunks per 32-elem row)
      const int ch = c & 3;
      const __bf16* g = A + (size_t)(m0b + r) * KDIM + k0 + ch * 8;
      const __bf16* l = &As[buf][r * PITCH_A + ch * 8];
      asm volatile("global_load_async_to_lds_b128 %0, %1, off"
                   :: "v"((unsigned)(size_t)l), "v"(g)
                   : "memory");
    }
    // B: 128x32 f32 = 1024 16B chunks, 4 per thread; convert to bf16 in LDS
#pragma unroll
    for (int q = 0; q < 4; ++q) {
      const int c  = q * 256 + tid;
      const int r  = c >> 3;        // row 0..127 (8 chunks per 32-elem row)
      const int ch = c & 7;
      int n = n0b + r;
      n = (n < NDIM) ? n : (NDIM - 1);   // tail clamp; never stored
      const f32x4 v = *(const f32x4*)(W + (size_t)n * KDIM + k0 + ch * 4);
      bf16x4 t;
      t[0] = (__bf16)v[0]; t[1] = (__bf16)v[1];
      t[2] = (__bf16)v[2]; t[3] = (__bf16)v[3];
      *(bf16x4*)&Bs[buf][r * PITCH_B + ch * 4] = t;   // ds_store_b64
    }
  };

  stage(0, 0);

  for (int it = 0; it < NSTEP; ++it) {
    if (it + 1 < NSTEP) {
      stage((it + 1) & 1, (it + 1) * BK);
      // 4 async loads just issued stay outstanding; current buffer's 4 retired
      asm volatile("s_wait_asynccnt 0x4" ::: "memory");
    } else {
      asm volatile("s_wait_asynccnt 0x0" ::: "memory");
    }
    __syncthreads();                       // all waves: current buffers ready

    const int buf = it & 1;

    // A fragments: 4 x (16m x 32k) from LDS, conflict-free ds_load_b128
    v16bf af[4];
#pragma unroll
    for (int i = 0; i < 4; ++i) {
      const __bf16* ap = &As[buf][(m0w + i * 16 + laneRow) * PITCH_A + laneHalf * 8];
      bf16x8* h = (bf16x8*)&af[i];
      h[0] = *(const bf16x8*)ap;           // K  0..7  (or 8..15)
      h[1] = *(const bf16x8*)(ap + 16);    // K 16..23 (or 24..31)
    }
#pragma unroll
    for (int j = 0; j < 4; ++j) {
      const __bf16* bp = &Bs[buf][(n0w + j * 16 + laneRow) * PITCH_B + laneHalf * 8];
      v16bf bv;
      bf16x8* h = (bf16x8*)&bv;
      h[0] = *(const bf16x8*)bp;
      h[1] = *(const bf16x8*)(bp + 16);
#pragma unroll
      for (int i = 0; i < 4; ++i) {
        acc[i][j] = __builtin_amdgcn_wmma_f32_16x16x32_bf16(
            false, af[i], false, bv, (short)0, acc[i][j], false, false);
      }
    }
    __syncthreads();                       // safe to overwrite buf next round
  }

  // ---- epilogue: bias add + guarded stores ----
#pragma unroll
  for (int j = 0; j < 4; ++j) {
    const int n = n0b + n0w + j * 16 + laneRow;
    if (n < NDIM) {
      const float bz = bias[n];
#pragma unroll
      for (int i = 0; i < 4; ++i) {
        const int mrow = m0b + m0w + i * 16 + laneHalf * 8;
#pragma unroll
        for (int r = 0; r < 8; ++r) {
          out[(size_t)(mrow + r) * NDIM + n] = acc[i][j][r] + bz;
        }
      }
    }
  }
}

// ---------------------------------------------------------------------------
extern "C" void kernel_launch(void* const* d_in, const int* in_sizes, int n_in,
                              void* d_out, int out_size, void* d_ws, size_t ws_size,
                              hipStream_t stream) {
  const long long* batch_data = (const long long*)d_in[0];
  const int*       ent2word   = (const int*)d_in[1];
  const float*     words_embd = (const float*)d_in[2];
  const float*     rel_embed  = (const float*)d_in[3];
  const float*     W          = (const float*)d_in[4];
  const float*     b          = (const float*)d_in[5];
  float*           out        = (float*)d_out;

  __bf16* A = (__bf16*)d_ws;   // 1024*768*2 = 1.5 MB of scratch

  prep_kernel<<<MDIM, HIDDEN, 0, stream>>>(batch_data, ent2word, words_embd,
                                           rel_embed, A);

  // grid.x = M tiles (4) adjacent so each W tile is reused from L2 ~4x
  dim3 grid(MDIM / BM, (NDIM + BN - 1) / BN);
  gemm_kernel<<<grid, 256, 0, stream>>>(A, W, b, out);
}
